// GemmaAttention_73933567033409
// MI455X (gfx1250) — compile-verified, matrix operand order checked by reading, not run on hardware
//
#include <hip/hip_runtime.h>
#include <hip/hip_bf16.h>

// ---------------------------------------------------------------------------
// GemmaAttention on MI455X (gfx1250, wave32):
//   - all matmuls via v_wmma_f32_16x16x32_f16
//   - global->LDS tile staging via GLOBAL_LOAD_ASYNC_TO_LDS_B128 (ASYNCcnt)
//     when the toolchain exposes the builtin; sync fallback otherwise
//   - double-buffered GEMM main loop (one barrier per K-step)
// ---------------------------------------------------------------------------

typedef __attribute__((ext_vector_type(16))) _Float16 v16h;
typedef __attribute__((ext_vector_type(8)))  _Float16 v8h;
typedef __attribute__((ext_vector_type(8)))  float    v8f;

#define B_   4
#define S_   2048
#define HID_ 2048
#define NH_  8
#define HD_  256
#define QKV_DIM_ 2560   // (NH + 2*NKV) * HD

// ---- async global->LDS support probe --------------------------------------
#if defined(__has_builtin)
#  if __has_builtin(__builtin_amdgcn_global_load_async_to_lds_b128)
#    define HAS_ASYNC_LDS 1
#  endif
#endif
#ifndef HAS_ASYNC_LDS
#  define HAS_ASYNC_LDS 0
#endif

#if HAS_ASYNC_LDS
// Builtin prototype (from hipcc diagnostic): pointee is int vector_size(16),
// src in address_space(1) (global), dst in address_space(3) (LDS).
typedef int v4i_gnu __attribute__((vector_size(4 * sizeof(int))));
typedef __attribute__((address_space(1))) v4i_gnu* as1_v4i;
typedef __attribute__((address_space(3))) v4i_gnu* as3_v4i;

__device__ __forceinline__ void async_copy_b128(const void* g, void* l) {
    // 16B per lane, global -> LDS, tracked by ASYNCcnt
    __builtin_amdgcn_global_load_async_to_lds_b128((as1_v4i)(g), (as3_v4i)(l), 0, 0);
}
__device__ __forceinline__ void wait_async0() {
#if __has_builtin(__builtin_amdgcn_s_wait_asynccnt)
    __builtin_amdgcn_s_wait_asynccnt(0);
#else
    asm volatile("s_wait_asynccnt 0x0" ::: "memory");
#endif
}
#endif

__device__ __forceinline__ v16h ld_frag(const _Float16* p) {
    // Lane fragment: 8 contiguous f16 + 8 contiguous f16 at +16 elements
    // (ISA 16-bit A/B layout: K 0..7 / 16..23 for lanes 0-15,
    //  K 8..15 / 24..31 for lanes 16-31 -- caller biases p by +8 for high lanes)
    v8h lo = *(const v8h*)(p);
    v8h hi = *(const v8h*)(p + 16);
    return __builtin_shufflevector(lo, hi, 0,1,2,3,4,5,6,7,8,9,10,11,12,13,14,15);
}

__device__ __forceinline__ v8f wmma16(v16h a, v16h b, v8f c) {
    return __builtin_amdgcn_wmma_f32_16x16x32_f16(false, a, false, b, (short)0, c, false, false);
}

// ---------------------------------------------------------------------------
// fp32 -> f16 conversion
// ---------------------------------------------------------------------------
__global__ __launch_bounds__(256) void cvt_f32_f16(const float* __restrict__ in,
                                                   _Float16* __restrict__ out, size_t n) {
    size_t i = (size_t)blockIdx.x * blockDim.x + threadIdx.x;
    size_t stride = (size_t)gridDim.x * blockDim.x;
    for (; i < n; i += stride) out[i] = (_Float16)in[i];
}

// ---------------------------------------------------------------------------
// WMMA GEMM: C[M,N] = A[M,K](f16) @ B[K,N](f16) + bias[N](f32)
// Block tile 128x128, 256 threads (8 waves, 2x4), BK=32, double-buffered LDS.
// ---------------------------------------------------------------------------
template <typename OutT>
__global__ __launch_bounds__(256) void wmma_gemm_bias(const _Float16* __restrict__ A,
                                                      const _Float16* __restrict__ Bm,
                                                      const float* __restrict__ bias,
                                                      OutT* __restrict__ C,
                                                      int M, int N, int K) {
    __shared__ _Float16 ldsA[2][128 * 48];   // [m][k], stride 48 halfs (96B, 16B-aligned)
    __shared__ _Float16 ldsB[2][128 * 48];   // [n][k] (transposed), stride 48 halfs

    const int tid  = threadIdx.x;
    const int lane = tid & 31;
    const int wid  = tid >> 5;
    const int wm   = wid & 1;        // 2 waves in M
    const int wn   = wid >> 1;       // 4 waves in N
    const int lcol = lane & 15;
    const int lhi  = lane >> 4;
    const int kb8  = lhi ? 8 : 0;

    const int n0 = blockIdx.x * 128;
    const int m0 = blockIdx.y * 128;

    v8f zero = {0.f,0.f,0.f,0.f,0.f,0.f,0.f,0.f};
    v8f acc[4][2];
#pragma unroll
    for (int mt = 0; mt < 4; ++mt)
#pragma unroll
        for (int nt = 0; nt < 2; ++nt) acc[mt][nt] = zero;

    // --- tile staging ------------------------------------------------------
    auto stage = [&](int buf, int kt) {
        // A tile 128x32 row-major: 512 chunks of 8 halfs (16B)
#pragma unroll
        for (int i = 0; i < 2; ++i) {
            int c = tid + i * 256;
            int row = c >> 2;
            int kc  = (c & 3) * 8;
            const _Float16* src = A + (size_t)(m0 + row) * K + kt + kc;
#if HAS_ASYNC_LDS
            async_copy_b128(src, &ldsA[buf][row * 48 + kc]);
#else
            v8h v = *(const v8h*)src;
            *(v8h*)&ldsA[buf][row * 48 + kc] = v;
#endif
        }
        // B tile 32x128, transposed into [n][k] (transpose => sync path)
#pragma unroll
        for (int i = 0; i < 2; ++i) {
            int c = tid + i * 256;
            int kr = c >> 4;
            int nc = (c & 15) * 8;
            v8h v = *(const v8h*)(Bm + (size_t)(kt + kr) * N + n0 + nc);
#pragma unroll
            for (int j = 0; j < 8; ++j) ldsB[buf][(nc + j) * 48 + kr] = v[j];
        }
    };

    const int nk = K >> 5;
    stage(0, 0);
#if HAS_ASYNC_LDS
    wait_async0();
#endif
    __syncthreads();

    for (int it = 0; it < nk; ++it) {
        const int cur = it & 1;
        // prefetch tile after next into L2
        if (it + 2 < nk) {
            __builtin_prefetch(A + (size_t)(m0 + (tid >> 2)) * K + (it + 2) * 32, 0, 1);
            __builtin_prefetch(Bm + (size_t)((it + 2) * 32 + (tid >> 4)) * N + n0, 0, 1);
        }
        // stage next tile into the other buffer; overlaps with WMMAs below
        if (it + 1 < nk) stage(cur ^ 1, (it + 1) * 32);

        v16h afrag[4], bfrag[2];
#pragma unroll
        for (int mt = 0; mt < 4; ++mt)
            afrag[mt] = ld_frag(&ldsA[cur][(wm * 64 + mt * 16 + lcol) * 48 + kb8]);
#pragma unroll
        for (int nt = 0; nt < 2; ++nt)
            bfrag[nt] = ld_frag(&ldsB[cur][(wn * 32 + nt * 16 + lcol) * 48 + kb8]);
#pragma unroll
        for (int mt = 0; mt < 4; ++mt)
#pragma unroll
            for (int nt = 0; nt < 2; ++nt)
                acc[mt][nt] = wmma16(afrag[mt], bfrag[nt], acc[mt][nt]);

#if HAS_ASYNC_LDS
        wait_async0();
#endif
        __syncthreads();   // single barrier per K-step (ping-pong buffers)
    }

    // epilogue: C layout row = r + 8*lhi, col = lcol
#pragma unroll
    for (int mt = 0; mt < 4; ++mt) {
#pragma unroll
        for (int nt = 0; nt < 2; ++nt) {
            int col = n0 + wn * 32 + nt * 16 + lcol;
            float bv = bias[col];
#pragma unroll
            for (int r = 0; r < 8; ++r) {
                int row = m0 + wm * 64 + mt * 16 + r + 8 * lhi;
                C[(size_t)row * N + col] = (OutT)(acc[mt][nt][r] + bv);
            }
        }
    }
}

// ---------------------------------------------------------------------------
// RoPE + head split: qkv[B*S, 2560] f16 -> q[B,NH,S,HD], k[B,S,HD], v[B,S,HD]
// ---------------------------------------------------------------------------
__global__ __launch_bounds__(256) void rope_split(const _Float16* __restrict__ qkv,
                                                  const float* __restrict__ fc,
                                                  _Float16* __restrict__ q,
                                                  _Float16* __restrict__ k,
                                                  _Float16* __restrict__ v) {
    const int r = blockIdx.x;           // b*S + s
    const int b = r >> 11;
    const int s = r & 2047;
    const _Float16* row = qkv + (size_t)r * QKV_DIM_;
    const float* f = fc + (size_t)s * HD_;   // [128 pairs][cos,sin]

    for (int idx = threadIdx.x; idx < NH_ * (HD_ / 2); idx += 256) {
        int h = idx >> 7, d2 = idx & 127;
        float c  = f[2 * d2], sn = f[2 * d2 + 1];
        float re = (float)row[h * HD_ + 2 * d2];
        float im = (float)row[h * HD_ + 2 * d2 + 1];
        size_t o = ((size_t)(b * NH_ + h) * S_ + s) * HD_ + 2 * d2;
        q[o]     = (_Float16)(re * c - im * sn);
        q[o + 1] = (_Float16)(re * sn + im * c);
    }
    if (threadIdx.x < HD_ / 2) {
        int d2 = threadIdx.x;
        float c  = f[2 * d2], sn = f[2 * d2 + 1];
        float re = (float)row[NH_ * HD_ + 2 * d2];
        float im = (float)row[NH_ * HD_ + 2 * d2 + 1];
        size_t o = (size_t)r * HD_ + 2 * d2;
        k[o]     = (_Float16)(re * c - im * sn);
        k[o + 1] = (_Float16)(re * sn + im * c);
    }
    v[(size_t)r * HD_ + threadIdx.x] = row[(NH_ + 1) * HD_ + threadIdx.x];
}

// ---------------------------------------------------------------------------
// Flash attention (causal, GQA NKV=1): 64 queries/block, 32-key tiles.
// 128 threads = 4 waves, each wave owns 16 query rows.
// Q fragments live in registers for the whole kernel (8 x v16h per wave).
// O accumulator: 16 x v8f per lane (16x256 per wave, fp32).
// K tile staged via async global->LDS; V via sync transpose stores.
// ---------------------------------------------------------------------------
#define BQ  64
#define BKV 32

__global__ __launch_bounds__(128) void flash_attn(const _Float16* __restrict__ qf,
                                                  const _Float16* __restrict__ kf,
                                                  const _Float16* __restrict__ vf,
                                                  _Float16* __restrict__ of) {
    __shared__ _Float16 sK[BKV * 272];    // [key][d], stride 272 (544B, 16B-aligned)
    __shared__ _Float16 sVt[HD_ * 48];    // [d][key], stride 48
    __shared__ float    sS[BQ * BKV];     // fp32 scores
    __shared__ _Float16 sP[BQ * 48];      // softmaxed probs, stride 48
    __shared__ float    sM[BQ], sL[BQ], sF[BQ];

    const int qb = blockIdx.x, h = blockIdx.y, b = blockIdx.z;
    const int tid  = threadIdx.x;
    const int lane = tid & 31, wid = tid >> 5;
    const int lcol = lane & 15, lhi = lane >> 4;
    const int kb8  = lhi ? 8 : 0;
    const int q0   = qb * BQ;

    const _Float16* qbase = qf + ((size_t)(b * NH_ + h) * S_ + q0) * HD_;
    const _Float16* kbase = kf + (size_t)b * S_ * HD_;
    const _Float16* vbase = vf + (size_t)b * S_ * HD_;

    const int rowA = wid * 16 + lcol;     // local query row for A fragments

    // Q fragments (K-dim = 256 -> 8 chunks of 32), kept in registers
    v16h qfrag[8];
#pragma unroll
    for (int kc = 0; kc < 8; ++kc)
        qfrag[kc] = ld_frag(qbase + (size_t)rowA * HD_ + kc * 32 + kb8);

    v8f zero = {0.f,0.f,0.f,0.f,0.f,0.f,0.f,0.f};
    v8f accO[16];
#pragma unroll
    for (int i = 0; i < 16; ++i) accO[i] = zero;

    if (tid < BQ) { sM[tid] = -1e30f; sL[tid] = 0.f; }

    const int nkb = (q0 + BQ) / BKV;      // causal: only key blocks <= query block
    for (int kb = 0; kb < nkb; ++kb) {
        const int k0 = kb * BKV;
        __syncthreads();
        // stage K tile 32x256 row-major (async path when available)
#pragma unroll
        for (int i = 0; i < 8; ++i) {
            int c = tid + i * 128;
            int r = c >> 5, col = (c & 31) * 8;
            const _Float16* src = kbase + (size_t)(k0 + r) * HD_ + col;
#if HAS_ASYNC_LDS
            async_copy_b128(src, &sK[r * 272 + col]);
#else
            v8h v = *(const v8h*)src;
            *(v8h*)&sK[r * 272 + col] = v;
#endif
        }
        // stage V tile transposed -> [d][key]
#pragma unroll
        for (int i = 0; i < 8; ++i) {
            int c = tid + i * 128;
            int r = c >> 5, col = (c & 31) * 8;
            v8h v = *(const v8h*)(vbase + (size_t)(k0 + r) * HD_ + col);
#pragma unroll
            for (int j = 0; j < 8; ++j) sVt[(col + j) * 48 + r] = v[j];
        }
#if HAS_ASYNC_LDS
        wait_async0();
#endif
        __syncthreads();

        // S = Q @ K^T  (16 queries x 32 keys per wave; 16 wmma)
        v8f accS[2];
        accS[0] = zero; accS[1] = zero;
#pragma unroll
        for (int kc = 0; kc < 8; ++kc) {
#pragma unroll
            for (int nt = 0; nt < 2; ++nt) {
                v16h bfrag = ld_frag(&sK[(nt * 16 + lcol) * 272 + kc * 32 + kb8]);
                accS[nt] = wmma16(qfrag[kc], bfrag, accS[nt]);
            }
        }
        // scale + causal mask, dump to LDS fp32
#pragma unroll
        for (int nt = 0; nt < 2; ++nt) {
#pragma unroll
            for (int r = 0; r < 8; ++r) {
                int ml   = wid * 16 + r + 8 * lhi;
                int kpos = k0 + nt * 16 + lcol;
                int qpos = q0 + ml;
                float sv = accS[nt][r] * 0.0625f;   // HD^-0.5 = 1/16
                if (kpos > qpos) sv = -1e9f;
                sS[ml * BKV + nt * 16 + lcol] = sv;
            }
        }
        __syncthreads();

        // online softmax per row (threads 0..63)
        if (tid < BQ) {
            int rr = tid;
            float mold = sM[rr], mnew = mold;
#pragma unroll
            for (int j = 0; j < BKV; ++j) mnew = fmaxf(mnew, sS[rr * BKV + j]);
            float fac = __expf(mold - mnew);
            float lsum = 0.f;
#pragma unroll
            for (int j = 0; j < BKV; ++j) {
                float p = __expf(sS[rr * BKV + j] - mnew);
                sP[rr * 48 + j] = (_Float16)p;
                lsum += p;
            }
            sM[rr] = mnew;
            sL[rr] = sL[rr] * fac + lsum;
            sF[rr] = fac;
        }
        __syncthreads();

        // rescale O accumulators
        float fr[8];
#pragma unroll
        for (int r = 0; r < 8; ++r) fr[r] = sF[wid * 16 + r + 8 * lhi];
#pragma unroll
        for (int nt = 0; nt < 16; ++nt)
#pragma unroll
            for (int r = 0; r < 8; ++r) accO[nt][r] *= fr[r];

        // O += P @ V  (16 wmma, K-dim = 32 keys)
        v16h pfrag = ld_frag(&sP[rowA * 48 + kb8]);
#pragma unroll
        for (int nt = 0; nt < 16; ++nt) {
            v16h bfrag = ld_frag(&sVt[(nt * 16 + lcol) * 48 + kb8]);
            accO[nt] = wmma16(pfrag, bfrag, accO[nt]);
        }
    }

    __syncthreads();
    float inv[8];
#pragma unroll
    for (int r = 0; r < 8; ++r) inv[r] = 1.f / sL[wid * 16 + r + 8 * lhi];
#pragma unroll
    for (int nt = 0; nt < 16; ++nt) {
#pragma unroll
        for (int r = 0; r < 8; ++r) {
            int ml = wid * 16 + r + 8 * lhi;
            size_t row = (size_t)(b * S_ + q0 + ml);
            of[row * (size_t)HID_ + h * HD_ + nt * 16 + lcol] =
                (_Float16)(accO[nt][r] * inv[r]);
        }
    }
}

// ---------------------------------------------------------------------------
// Launch
// ---------------------------------------------------------------------------
extern "C" void kernel_launch(void* const* d_in, const int* in_sizes, int n_in,
                              void* d_out, int out_size, void* d_ws, size_t ws_size,
                              hipStream_t stream) {
    (void)in_sizes; (void)n_in; (void)out_size; (void)ws_size;
    const float* x      = (const float*)d_in[0];
    const float* w_qkv  = (const float*)d_in[1];
    const float* b_qkv  = (const float*)d_in[2];
    const float* w_o    = (const float*)d_in[3];
    const float* b_o    = (const float*)d_in[4];
    const float* freqs  = (const float*)d_in[5];
    // d_in[6] (mask) unused: causal mask computed from indices in-kernel.
    float* out = (float*)d_out;

    char* w = (char*)d_ws;
    size_t o = 0;
    _Float16* xf    = (_Float16*)(w + o); o += (size_t)B_*S_*HID_ * 2;       // 32 MB
    _Float16* wqkvf = (_Float16*)(w + o); o += (size_t)HID_*QKV_DIM_ * 2;    // 10 MB
    _Float16* wof   = (_Float16*)(w + o); o += (size_t)HID_*HID_ * 2;        // 8 MB
    _Float16* qkvf  = (_Float16*)(w + o); o += (size_t)B_*S_*QKV_DIM_ * 2;   // 40 MB
    _Float16* qhf   = (_Float16*)(w + o); o += (size_t)B_*NH_*S_*HD_ * 2;    // 32 MB
    _Float16* khf   = (_Float16*)(w + o); o += (size_t)B_*S_*HD_ * 2;        // 4 MB
    _Float16* vhf   = (_Float16*)(w + o); o += (size_t)B_*S_*HD_ * 2;        // 4 MB
    _Float16* aof   = (_Float16*)(w + o); o += (size_t)B_*S_*HID_ * 2;       // 32 MB

    cvt_f32_f16<<<4096, 256, 0, stream>>>(x,     xf,    (size_t)B_*S_*HID_);
    cvt_f32_f16<<<2048, 256, 0, stream>>>(w_qkv, wqkvf, (size_t)HID_*QKV_DIM_);
    cvt_f32_f16<<<2048, 256, 0, stream>>>(w_o,   wof,   (size_t)HID_*HID_);

    // QKV projection: [8192,2048] @ [2048,2560] + bias -> f16
    wmma_gemm_bias<_Float16><<<dim3(QKV_DIM_/128, (B_*S_)/128), 256, 0, stream>>>(
        xf, wqkvf, b_qkv, qkvf, B_*S_, QKV_DIM_, HID_);

    rope_split<<<B_*S_, 256, 0, stream>>>(qkvf, freqs, qhf, khf, vhf);

    flash_attn<<<dim3(S_/BQ, NH_, B_), 128, 0, stream>>>(qhf, khf, vhf, aof);

    // Output projection: [8192,2048] @ [2048,2048] + bias -> f32 d_out
    wmma_gemm_bias<float><<<dim3(HID_/128, (B_*S_)/128), 256, 0, stream>>>(
        aof, wof, b_o, out, B_*S_, HID_, HID_);
}
